// LineA_76347338653808
// MI455X (gfx1250) — compile-verified, matrix operand order checked by reading, not run on hardware
//
#include <hip/hip_runtime.h>
#include <hip/hip_bf16.h>

// Problem constants (match reference)
#define BB   128
#define TT   1024
#define FF   64
#define HH   256
#define G4H  (4*HH)
#define NWG  16          // persistent workgroups, each owns 16 H-columns
#define NTH  256         // 8 wave32 per WG
#define ALPHA 0.3f

typedef __bf16 bf16_t;
typedef __attribute__((ext_vector_type(16))) __bf16 v16bf;
typedef __attribute__((ext_vector_type(8)))  __bf16 v8bf;
typedef __attribute__((ext_vector_type(8)))  float  v8f;

// LDS pitches (elements). 264*2B = 528B = 33 dwords*16 -> 16B aligned rows,
// and col-to-col stride of 132 dwords (%64 = 4 banks) => conflict-light b128 reads.
#define PR 264   // recurrent Bt pitch (256 + 8)
#define PX 72    // input-kernel Bt pitch (64 + 8)

// ---- helpers -------------------------------------------------------------

__device__ __forceinline__ v16bf load16(const bf16_t* p0, const bf16_t* p1) {
  // Two contiguous 16B chunks -> one 16-bit A/B WMMA fragment per lane
  v8bf a = *(const v8bf*)p0;
  v8bf b = *(const v8bf*)p1;
  v16bf r;
#pragma unroll
  for (int i = 0; i < 8; ++i) { r[i] = a[i]; r[i + 8] = b[i]; }
  return r;
}

__device__ __forceinline__ float sigmoid_f(float x) {
  return 1.0f / (1.0f + __expf(-x));
}
__device__ __forceinline__ float tanh_f(float x) {
  // 1 - 2/(e^{2x}+1): saturates cleanly at +/-1
  return 1.0f - 2.0f / (__expf(2.0f * x) + 1.0f);
}

// device-wide split barrier over NWG persistent workgroups
__device__ __forceinline__ void grid_barrier(unsigned* cnt, unsigned target) {
  __threadfence();               // release our h stores to device scope
  __syncthreads();
  if (threadIdx.x == 0) {
    __hip_atomic_fetch_add(cnt, 1u, __ATOMIC_RELEASE, __HIP_MEMORY_SCOPE_AGENT);
    while (__hip_atomic_load(cnt, __ATOMIC_ACQUIRE, __HIP_MEMORY_SCOPE_AGENT) < target) {
      __builtin_amdgcn_s_sleep(2);
    }
  }
  __syncthreads();
  __threadfence();               // acquire: invalidate L0 so h loads see fresh data
}

// ---- kernel 1: transpose/convert inputs to [T][B*F] bf16, reset barrier --

__global__ __launch_bounds__(NTH) void prep_kernel(const float* __restrict__ inp,
                                                   bf16_t* __restrict__ xbf,
                                                   unsigned* __restrict__ cnt) {
  if (blockIdx.x == 0 && threadIdx.x == 0) cnt[0] = 0u;   // graph-replay safe reset
  const int total = BB * TT * FF;
  for (int i = blockIdx.x * blockDim.x + threadIdx.x; i < total;
       i += gridDim.x * blockDim.x) {
    int b = i / (TT * FF);
    int r = i % (TT * FF);
    int t = r / FF;
    int f = r % FF;
    xbf[(size_t)t * (BB * FF) + b * FF + f] = (bf16_t)inp[i];
  }
}

// ---- kernel 2: persistent fused LSTM scan --------------------------------

__global__ __launch_bounds__(NTH) void lstm_scan_kernel(
    const bf16_t* __restrict__ xbf,    // [T][B*F] bf16
    const float*  __restrict__ rker,   // [H][4H] fp32
    const float*  __restrict__ xker,   // [F][4H] fp32
    const float*  __restrict__ bias,   // [4H]
    bf16_t* __restrict__ hbuf0,        // [B][H] bf16 ping
    bf16_t* __restrict__ hbuf1,        // [B][H] bf16 pong
    unsigned* __restrict__ cnt,
    float* __restrict__ out)           // [B][H] fp32
{
  __shared__ __align__(16) bf16_t sBr[64 * PR];   // recurrent slice, column-major
  __shared__ __align__(16) bf16_t sBx[64 * PX];   // input-kernel slice, column-major

  const int g    = blockIdx.x;        // which 16-column H chunk we own
  const int tid  = threadIdx.x;
  const int wave = tid >> 5;          // 8 waves -> 8 row tiles of 16
  const int lane = tid & 31;
  const int l16  = lane & 15;
  const int hb   = (lane < 16) ? 0 : 8;
  const int m0   = wave * 16;

  // Stage weight slices into LDS as Bt[col][k] (bf16)
  for (int idx = tid; idx < 64 * HH; idx += NTH) {
    int c = idx >> 8, k = idx & 255;
    int gcol = (c >> 4) * HH + g * 16 + (c & 15);        // gate*H + ourcol
    sBr[c * PR + k] = (bf16_t)rker[(size_t)k * G4H + gcol];
  }
  for (int idx = tid; idx < 64 * FF; idx += NTH) {
    int c = idx >> 6, k = idx & 63;
    int gcol = (c >> 4) * HH + g * 16 + (c & 15);
    sBx[c * PX + k] = (bf16_t)xker[(size_t)k * G4H + gcol];
  }

  const int colH = g * 16 + l16;      // our H column for this lane
  const float bi = bias[0 * HH + colH];
  const float bf_ = bias[1 * HH + colH];
  const float bc = bias[2 * HH + colH];
  const float bo = bias[3 * HH + colH];

  // h0 = 0 in ping buffer (we own columns [g*16, g*16+16))
  for (int idx = tid; idx < BB * 16; idx += NTH) {
    int row = idx >> 4, cc = idx & 15;
    hbuf0[row * HH + g * 16 + cc] = (bf16_t)0.0f;
  }

  float c_reg[8];
#pragma unroll
  for (int r = 0; r < 8; ++r) c_reg[r] = 0.0f;

  unsigned epoch = 0;
  grid_barrier(cnt, (++epoch) * NWG);   // h0 visible everywhere

  for (int t = 0; t < TT; ++t) {
    const bf16_t* __restrict__ hcur = (t & 1) ? hbuf1 : hbuf0;
    bf16_t*       __restrict__ hnxt = (t & 1) ? hbuf0 : hbuf1;

    v8f acc[4];
#pragma unroll
    for (int r = 0; r < 8; ++r) { acc[0][r] = bi; acc[1][r] = bf_; acc[2][r] = bc; acc[3][r] = bo; }

    // --- input projection part: K = 64 (A from transposed bf16 inputs) ---
    const bf16_t* xrow = xbf + (size_t)t * (BB * FF);
#pragma unroll
    for (int kk = 0; kk < 2; ++kk) {
      const int kbase = kk * 32 + hb;
      const bf16_t* pa = xrow + (m0 + l16) * FF + kbase;
      v16bf A = load16(pa, pa + 16);
#pragma unroll
      for (int n = 0; n < 4; ++n) {
        const bf16_t* pb = sBx + (n * 16 + l16) * PX + kbase;
        v16bf Bv = load16(pb, pb + 16);
        acc[n] = __builtin_amdgcn_wmma_f32_16x16x32_bf16(
            false, A, false, Bv, (short)0, acc[n], false, false);
      }
    }

    // --- recurrent part: K = 256 (A = h from L2-resident ping buffer) ---
#pragma unroll
    for (int kk = 0; kk < 8; ++kk) {
      const int kbase = kk * 32 + hb;
      const bf16_t* pa = hcur + (m0 + l16) * HH + kbase;
      v16bf A = load16(pa, pa + 16);
#pragma unroll
      for (int n = 0; n < 4; ++n) {
        const bf16_t* pb = sBr + (n * 16 + l16) * PR + kbase;
        v16bf Bv = load16(pb, pb + 16);
        acc[n] = __builtin_amdgcn_wmma_f32_16x16x32_bf16(
            false, A, false, Bv, (short)0, acc[n], false, false);
      }
    }

    // --- gates + state update (C layout: VGPR r -> row m0+r+hb, col colH) ---
#pragma unroll
    for (int r = 0; r < 8; ++r) {
      float ig = sigmoid_f(acc[0][r]);
      float fg = sigmoid_f(acc[1][r]);
      float ct = tanh_f(acc[2][r]);
      float og = sigmoid_f(acc[3][r]);
      float cn = fg * c_reg[r] + ig * ct;
      c_reg[r] = cn;
      float hn = og * tanh_f(cn);
      const int row = m0 + r + hb;
      if (t == TT - 1) {
        out[row * HH + colH] = (hn >= 0.0f) ? hn : ALPHA * hn;   // LeakyReLU(0.3)
      } else {
        hnxt[row * HH + colH] = (bf16_t)hn;
      }
    }

    if (t < TT - 1) grid_barrier(cnt, (++epoch) * NWG);
  }
}

// ---- host entry ----------------------------------------------------------

extern "C" void kernel_launch(void* const* d_in, const int* in_sizes, int n_in,
                              void* d_out, int out_size, void* d_ws, size_t ws_size,
                              hipStream_t stream) {
  const float* inputs = (const float*)d_in[0];   // [B,T,F]
  const float* xker   = (const float*)d_in[1];   // [F,4H]
  const float* rker   = (const float*)d_in[2];   // [H,4H]
  const float* bias   = (const float*)d_in[3];   // [4H]

  char* ws = (char*)d_ws;
  unsigned* cnt = (unsigned*)ws;                          // barrier counter
  bf16_t* h0   = (bf16_t*)(ws + 1024);                    // 64 KB ping
  bf16_t* h1   = (bf16_t*)(ws + 1024 + 64 * 1024);        // 64 KB pong
  bf16_t* xbf  = (bf16_t*)(ws + 1024 + 128 * 1024);       // 16 MB [T][B*F] bf16

  prep_kernel<<<4096, NTH, 0, stream>>>(inputs, xbf, cnt);
  lstm_scan_kernel<<<NWG, NTH, 0, stream>>>(xbf, rker, xker, bias,
                                            h0, h1, cnt, (float*)d_out);
}
